// MultiHeadAttention_33432025432186
// MI455X (gfx1250) — matrix operand order repaired and checked by reading, not compiled, not run on hardware
//
#include <hip/hip_runtime.h>

// MHA forward for MI455X (gfx1250, wave32, WMMA 16x16x32 f16, TDM staging).
// Pipeline: transpose weights->f16 | GEMM projections (Q,K,V; K stored [B,H,dk,S])
//           | flash attention (online softmax) | output GEMM -> fp32 d_out.
// Workspace: 4*1M (W^T) + 3*8M (Qp/Kp/Vp) + 8M (ctx) f16 = 72 MB.

#define D_MODEL 1024
#define NHEADS  16
#define DKH     64
#define BATCH   4
#define SEQ     2048
#define ROWS    (BATCH * SEQ)   // 8192

typedef _Float16 h16 __attribute__((ext_vector_type(16)));
typedef _Float16 h8  __attribute__((ext_vector_type(8)));
typedef float    f8  __attribute__((ext_vector_type(8)));
typedef unsigned int u32x4 __attribute__((ext_vector_type(4)));
typedef int          i32x4 __attribute__((ext_vector_type(4)));
typedef int          i32x8 __attribute__((ext_vector_type(8)));

#if defined(__HIP_DEVICE_COMPILE__) && \
    __has_builtin(__builtin_amdgcn_tensor_load_to_lds) && \
    __has_builtin(__builtin_amdgcn_s_wait_tensorcnt)
#define USE_TDM 1
#else
#define USE_TDM 0
#endif

static __device__ __forceinline__ f8 zero_f8() {
  f8 z;
#pragma unroll
  for (int i = 0; i < 8; ++i) z[i] = 0.0f;
  return z;
}

static __device__ __forceinline__ f8 wmma_f16(h16 a, h16 b, f8 c) {
  // (neg_a, A, neg_b, B, c_mod, C, reuse_a, reuse_b)
  return __builtin_amdgcn_wmma_f32_16x16x32_f16(false, a, false, b, (short)0, c,
                                                false, false);
}

// A fragment (16x32 f16, MxK) from row-major source (ld in elements), K window k0.
// ISA layout: lane L -> row L%16; lanes 0-15 hold K=[0..7],[16..23], lanes 16-31
// hold K=[8..15],[24..31] (v16h elems 0..7 = first chunk, 8..15 = second).
static __device__ __forceinline__ h16 load_a_frag(const _Float16* base, int ld,
                                                  int k0, int lane) {
  const int row = lane & 15;
  const int c   = lane >> 4;
  const _Float16* p = base + (size_t)row * ld + k0 + c * 8;
  h8 lo = *(const h8*)(p);
  h8 hi = *(const h8*)(p + 16);
  h16 f;
#pragma unroll
  for (int i = 0; i < 8; ++i) { f[i] = lo[i]; f[8 + i] = hi[i]; }
  return f;
}

// Same A fragment but converting from fp32 source on the fly.
static __device__ __forceinline__ h16 load_a_frag_f32(const float* base, int ld,
                                                      int k0, int lane) {
  const int row = lane & 15;
  const int c   = lane >> 4;
  const float* p = base + (size_t)row * ld + k0 + c * 8;
  f8 lo = *(const f8*)(p);
  f8 hi = *(const f8*)(p + 16);
  h16 f;
#pragma unroll
  for (int i = 0; i < 8; ++i) {
    f[i]     = (_Float16)lo[i];
    f[8 + i] = (_Float16)hi[i];
  }
  return f;
}

// B fragment (32x16 f16, KxN): lane L holds column col0+L%16; elems 0..7 = K 8*(L/16)+i,
// elems 8..15 = K 16+8*(L/16)+i (same K order as the A fragment).
static __device__ __forceinline__ h16 load_b_frag(const _Float16* base, int ld,
                                                  int row0, int col0, int lane) {
  const _Float16* p = base + (size_t)(row0 + (lane >> 4) * 8) * ld + col0 + (lane & 15);
  h16 f;
#pragma unroll
  for (int i = 0; i < 8; ++i) {
    f[i]     = p[(size_t)i * ld];
    f[8 + i] = p[(size_t)(16 + i) * ld];
  }
  return f;
}

#if USE_TDM
// TDM: DMA a 32x128 f16 tile of WT (row-major, ld=D_MODEL) starting at
// (k0, n0) into LDS at lds_tile, with hardware padding of 8 DWORDs after
// every 64 DWORDs (row = 128 f16 = 256B) -> LDS row stride 288B (144 halves).
// D# fields per CDNA5 ISA. This toolchain's
// builtin is the 6-arg form: (g0 u32x4, g1 i32x8, g2 i32x4, g3 i32x4,
// i32x8, i32 cpol).
static __device__ __forceinline__ void tdm_load_wt(_Float16* lds_tile,
                                                   const _Float16* WT,
                                                   int k0, int n0) {
  const unsigned long long ga =
      (unsigned long long)(uintptr_t)(WT + (size_t)k0 * D_MODEL + n0);
  const unsigned int lds_addr = (unsigned int)(uintptr_t)lds_tile;  // low 32 = LDS offset

  u32x4 g0;
  g0[0] = 1u;                                   // count=1 (valid user descriptor)
  g0[1] = lds_addr;                             // lds_addr [63:32]
  g0[2] = (unsigned int)ga;                     // global_addr [95:64]
  g0[3] = (unsigned int)((ga >> 32) & 0x01ffffffu) | (2u << 30);  // addr hi | type=2

  i32x8 g1;
  // [17:16] data_size=1 (2B), [20] pad_enable, [24:22] pad_interval=5 (64 DW),
  // [31:25] pad_amount=7 (8 DW); workgroup_mask=0 (not in cluster).
  g1[0] = (int)((1u << 16) | (1u << 20) | (5u << 22) | (7u << 25));
  g1[1] = (int)(((unsigned)D_MODEL & 0xffffu) << 16);        // tensor_dim0[15:0]
  g1[2] = (int)((((unsigned)D_MODEL >> 16) & 0xffffu) |
                (((unsigned)D_MODEL & 0xffffu) << 16));      // dim0 hi | dim1 lo
  g1[3] = (int)((((unsigned)D_MODEL >> 16) & 0xffffu) |
                (128u << 16));                               // dim1 hi | tile_dim0=128
  g1[4] = (int)(32u);                                        // tile_dim1=32, tile_dim2=0
  g1[5] = (int)((unsigned)D_MODEL);                          // dim0_stride[31:0]
  g1[6] = 0;                                                 // stride hi | dim1_stride lo
  g1[7] = 0;

  i32x4 gz4;
  gz4[0] = 0; gz4[1] = 0; gz4[2] = 0; gz4[3] = 0;            // 2D: groups 2/3 unused
  i32x8 gz8;
#pragma unroll
  for (int i = 0; i < 8; ++i) gz8[i] = 0;

  __builtin_amdgcn_tensor_load_to_lds(g0, g1, gz4, gz4, gz8, 0);
}
#endif

// ---------------------------------------------------------------------------
// W (fp32 [M,D]) -> W^T (f16 [D,M]) so GEMM B-tiles are K-major contiguous.
__global__ void __launch_bounds__(256) transpose_f16(const float* __restrict__ in,
                                                     _Float16* __restrict__ out) {
  const int i = blockIdx.x * blockDim.x + threadIdx.x;  // index into out [D, M]
  const int d = i >> 10;
  const int m = i & (D_MODEL - 1);
  out[i] = (_Float16)in[(size_t)m * D_MODEL + d];
}

// ---------------------------------------------------------------------------
// C[m,n] = sum_k A[m,k] * WT[k,n] + bias[n]
// MODE 0: A=f32, out f16 scattered to [B,H,S,DKH]   (Q / V projection)
// MODE 1: A=f32, out f16 scattered to [B,H,DKH,S]   (K projection, transposed)
// MODE 3: A=f16, out f32 row-major [ROWS, D_MODEL]  (final output GEMM)
// W tiles are staged into ping-pong LDS buffers by the Tensor Data Mover
// (wave 0 DMAs tile k+1 while all 8 waves consume tile k). The k-loop is
// hand-unrolled x2 so each copy uses a compile-time LDS buffer index.
template <int MODE>
__global__ void __launch_bounds__(256) gemm_xwt(const void* __restrict__ Ain,
                                                const _Float16* __restrict__ WT,
                                                const float* __restrict__ bias,
                                                void* __restrict__ out) {
  constexpr int LDSB = 144;  // padded stride (halves): 288B rows, 32B aligned
  __shared__ _Float16 ldsB[2][32 * LDSB];

  const int tid  = threadIdx.x;
  const int wave = tid >> 5;
  const int lane = tid & 31;
  const int n0 = blockIdx.x * 128;
  const int m0 = blockIdx.y * 128 + wave * 16;

  f8 acc[8];
#pragma unroll
  for (int j = 0; j < 8; ++j) acc[j] = zero_f8();

#if USE_TDM
  if (wave == 0) tdm_load_wt(&ldsB[0][0], WT, 0, n0);
#endif

  // One ping-pong step: stage (or wait for) tile at k0 in buffer BUF,
  // prefetch tile k0+32 into BUF^1, multiply-accumulate from BUF.
  auto step = [&](int k0, int buf_ci) __attribute__((always_inline)) {
#if USE_TDM
    if (wave == 0) {
      if (k0 + 32 < D_MODEL) {
        tdm_load_wt(&ldsB[buf_ci ^ 1][0], WT, k0 + 32, n0);  // prefetch next tile
        __builtin_amdgcn_s_wait_tensorcnt(1);                // current tile landed
      } else {
        __builtin_amdgcn_s_wait_tensorcnt(0);
      }
    }
#else
    {  // fallback: cooperative vector staging of WT[k0:k0+32, n0:n0+128]
      const int r  = tid >> 3;  // 0..31 (k row)
      const int cc = tid & 7;   // 0..7  (16-half column chunk)
      *(h16*)(&ldsB[buf_ci][r * LDSB + cc * 16]) =
          *(const h16*)(WT + (size_t)(k0 + r) * D_MODEL + n0 + cc * 16);
    }
#endif
    __syncthreads();

    h16 afrag;
    if (MODE == 3)
      afrag = load_a_frag((const _Float16*)Ain + (size_t)m0 * D_MODEL, D_MODEL,
                          k0, lane);
    else
      afrag = load_a_frag_f32((const float*)Ain + (size_t)m0 * D_MODEL, D_MODEL,
                              k0, lane);

    if (k0 + 32 < D_MODEL) {  // prefetch next A chunk (global_prefetch_b8)
      const size_t esz = (MODE == 3) ? 2 : 4;
      __builtin_prefetch((const char*)Ain +
                             ((size_t)(m0 + (lane & 15)) * D_MODEL + k0 + 32) * esz,
                         0, 1);
    }

    const _Float16* brow = &ldsB[buf_ci][(lane >> 4) * 8 * LDSB + (lane & 15)];
#pragma unroll
    for (int j = 0; j < 8; ++j) {
      h16 bfrag;
#pragma unroll
      for (int i = 0; i < 8; ++i) {
        bfrag[i]     = brow[i * LDSB + j * 16];
        bfrag[8 + i] = brow[(16 + i) * LDSB + j * 16];
      }
      acc[j] = wmma_f16(afrag, bfrag, acc[j]);
    }
    __syncthreads();
  };

  for (int k0 = 0; k0 < D_MODEL; k0 += 64) {  // x2 unroll: constant buffer ids
    step(k0, 0);
    step(k0 + 32, 1);
  }

  // Epilogue. C/D layout: VGPR r, lanes 0-15 -> M=r, lanes 16-31 -> M=r+8; N=lane%16.
  const int group = lane >> 4;
  const int nl    = lane & 15;
#pragma unroll
  for (int j = 0; j < 8; ++j) {
    const int n = n0 + j * 16 + nl;
    const float bv = bias[n];
#pragma unroll
    for (int r = 0; r < 8; ++r) {
      const int m = m0 + r + 8 * group;
      const float val = acc[j][r] + bv;
      if (MODE == 3) {
        ((float*)out)[(size_t)m * D_MODEL + n] = val;
      } else {
        const int b  = m / SEQ;
        const int s  = m % SEQ;
        const int h  = n / DKH;
        const int dk = n % DKH;
        size_t idx;
        if (MODE == 1)
          idx = (((size_t)b * NHEADS + h) * DKH + dk) * SEQ + s;   // [B,H,dk,S]
        else
          idx = (((size_t)b * NHEADS + h) * SEQ + s) * DKH + dk;   // [B,H,S,dk]
        ((_Float16*)out)[idx] = (_Float16)val;
      }
    }
  }
}

// ---------------------------------------------------------------------------
// Causal flash attention. One wave owns a 16-row q strip; 8 waves/block cover
// 128 q rows of one (b,h). 32 keys per step: QK^T (4 wmma) -> online softmax
// -> P via per-wave LDS tile -> PV (4 wmma).
__global__ void __launch_bounds__(256) flash_attn(const _Float16* __restrict__ Qp,
                                                  const _Float16* __restrict__ Kp,
                                                  const _Float16* __restrict__ Vp,
                                                  _Float16* __restrict__ ctx) {
  __shared__ _Float16 pbuf[8][16 * 32];  // per-wave 16x32 P tile (1KB each)

  const int bh    = blockIdx.y;
  const int b     = bh >> 4;
  const int h     = bh & (NHEADS - 1);
  const int wave  = threadIdx.x >> 5;
  const int lane  = threadIdx.x & 31;
  const int group = lane >> 4;
  const int nl    = lane & 15;
  const int q0    = blockIdx.x * 128 + wave * 16;

  const _Float16* Qb = Qp + (size_t)bh * SEQ * DKH;  // [S, dk]
  const _Float16* Kb = Kp + (size_t)bh * DKH * SEQ;  // [dk, S]
  const _Float16* Vb = Vp + (size_t)bh * SEQ * DKH;  // [S, dk]

  const h16 aq0 = load_a_frag(Qb + (size_t)q0 * DKH, DKH, 0, lane);
  const h16 aq1 = load_a_frag(Qb + (size_t)q0 * DKH, DKH, 32, lane);

  f8 cacc[4];
#pragma unroll
  for (int j = 0; j < 4; ++j) cacc[j] = zero_f8();

  float mrow[8], lrow[8];
#pragma unroll
  for (int r = 0; r < 8; ++r) { mrow[r] = -3.0e38f; lrow[r] = 0.0f; }

  _Float16* pw = &pbuf[wave][0];
  const float scale = 0.125f;  // 1/sqrt(64)

  for (int k0 = 0; k0 < q0 + 16; k0 += 32) {
    // scores: two 16x16 key tiles (keys k0..k0+15, k0+16..k0+31), dk = 64 = 2 chunks
    f8 s0 = zero_f8(), s1 = zero_f8();
    {
      h16 bk;
      bk = load_b_frag(Kb, SEQ, 0,  k0,      lane); s0 = wmma_f16(aq0, bk, s0);
      bk = load_b_frag(Kb, SEQ, 32, k0,      lane); s0 = wmma_f16(aq1, bk, s0);
      bk = load_b_frag(Kb, SEQ, 0,  k0 + 16, lane); s1 = wmma_f16(aq0, bk, s1);
      bk = load_b_frag(Kb, SEQ, 32, k0 + 16, lane); s1 = wmma_f16(aq1, bk, s1);
    }

    // scale + causal mask + online softmax (row = r + 8*group, key = k0+t*16+nl)
#pragma unroll
    for (int r = 0; r < 8; ++r) {
      const int qpos = q0 + r + 8 * group;
      float v0 = s0[r] * scale;
      float v1 = s1[r] * scale;
      if (k0 + nl      > qpos) v0 = -1.0e9f;
      if (k0 + 16 + nl > qpos) v1 = -1.0e9f;

      float mx = fmaxf(v0, v1);
#pragma unroll
      for (int off = 1; off < 16; off <<= 1)
        mx = fmaxf(mx, __shfl_xor(mx, off, 32));
      const float mnew = fmaxf(mrow[r], mx);

      const float p0 = __expf(v0 - mnew);
      const float p1 = __expf(v1 - mnew);
      float rs = p0 + p1;
#pragma unroll
      for (int off = 1; off < 16; off <<= 1) rs += __shfl_xor(rs, off, 32);

      const float alpha = __expf(mrow[r] - mnew);
      lrow[r] = lrow[r] * alpha + rs;
      mrow[r] = mnew;
#pragma unroll
      for (int j = 0; j < 4; ++j) cacc[j][r] *= alpha;

      const int row = r + 8 * group;
      pw[row * 32 + nl]      = (_Float16)p0;
      pw[row * 32 + 16 + nl] = (_Float16)p1;
    }

    // Same-wave LDS ops are in-order: reload P tile as 16x32 A fragment.
    const h16 ap = load_a_frag(pw, 32, 0, lane);
#pragma unroll
    for (int j = 0; j < 4; ++j) {
      h16 bv = load_b_frag(Vb, DKH, k0, j * 16, lane);
      cacc[j] = wmma_f16(ap, bv, cacc[j]);
    }
  }

  // Normalize and write ctx[b, qpos, h*64 + dk] (f16, head-interleaved).
#pragma unroll
  for (int j = 0; j < 4; ++j) {
#pragma unroll
    for (int r = 0; r < 8; ++r) {
      const int qpos = q0 + r + 8 * group;
      const float val = cacc[j][r] / lrow[r];
      ctx[((size_t)b * SEQ + qpos) * D_MODEL + h * DKH + j * 16 + nl] =
          (_Float16)val;
    }
  }
}

// ---------------------------------------------------------------------------
extern "C" void kernel_launch(void* const* d_in, const int* in_sizes, int n_in,
                              void* d_out, int out_size, void* d_ws, size_t ws_size,
                              hipStream_t stream) {
  (void)in_sizes; (void)n_in; (void)out_size; (void)ws_size;
  const float* q  = (const float*)d_in[0];
  const float* k  = (const float*)d_in[1];
  const float* v  = (const float*)d_in[2];
  // d_in[3] = causal mask (hardcoded in flash_attn)
  const float* Wq = (const float*)d_in[4];
  const float* bq = (const float*)d_in[5];
  const float* Wk = (const float*)d_in[6];
  const float* bk = (const float*)d_in[7];
  const float* Wv = (const float*)d_in[8];
  const float* bv = (const float*)d_in[9];
  const float* Wo = (const float*)d_in[10];
  const float* bo = (const float*)d_in[11];

  _Float16* ws = (_Float16*)d_ws;
  const size_t WE = (size_t)D_MODEL * D_MODEL;  // 1M elems
  const size_t NE = (size_t)ROWS * D_MODEL;     // 8M elems
  _Float16* WqT = ws;
  _Float16* WkT = WqT + WE;
  _Float16* WvT = WkT + WE;
  _Float16* WoT = WvT + WE;
  _Float16* Qp  = WoT + WE;
  _Float16* Kp  = Qp + NE;
  _Float16* Vp  = Kp + NE;
  _Float16* ctx = Vp + NE;   // total 36M f16 = 72 MB

  transpose_f16<<<(int)(WE / 256), 256, 0, stream>>>(Wq, WqT);
  transpose_f16<<<(int)(WE / 256), 256, 0, stream>>>(Wk, WkT);
  transpose_f16<<<(int)(WE / 256), 256, 0, stream>>>(Wv, WvT);
  transpose_f16<<<(int)(WE / 256), 256, 0, stream>>>(Wo, WoT);

  dim3 gg(D_MODEL / 128, ROWS / 128);  // (8, 64)
  gemm_xwt<0><<<gg, 256, 0, stream>>>((const void*)q, WqT, bq, (void*)Qp);
  gemm_xwt<1><<<gg, 256, 0, stream>>>((const void*)k, WkT, bk, (void*)Kp);
  gemm_xwt<0><<<gg, 256, 0, stream>>>((const void*)v, WvT, bv, (void*)Vp);

  dim3 ga(SEQ / 128, BATCH * NHEADS);  // (16, 64)
  flash_attn<<<ga, 256, 0, stream>>>(Qp, Kp, Vp, ctx);

  gemm_xwt<3><<<gg, 256, 0, stream>>>((const void*)ctx, WoT, bo, d_out);
}
